// Local_enhancer_62801011802556
// MI455X (gfx1250) — matrix operand order repaired and checked by reading, not compiled, not run on hardware
//
#include <hip/hip_runtime.h>
#include <hip/hip_bf16.h>
#include <cstdint>

typedef __bf16 bf16_t;
typedef __attribute__((ext_vector_type(16))) __bf16 v16bf;
typedef __attribute__((ext_vector_type(8)))  __bf16 v8bf;
typedef __attribute__((ext_vector_type(8)))  float  v8f;

// ---------------- layout conversion (LDS-tiled transposes, C == 32) ----------------
// fp32 NCDHW [32][V] -> bf16 voxel-major [V][32]; coalesced both sides.
__global__ void __launch_bounds__(256)
k_nchw_to_vc32(const float* __restrict__ in, bf16_t* __restrict__ out, int V) {
  __shared__ bf16_t t[256][33];
  int v0 = blockIdx.x * 256;
  int tid = threadIdx.x;
#pragma unroll
  for (int c = 0; c < 32; ++c)
    t[tid][c] = (bf16_t)in[(size_t)c * V + v0 + tid];      // coalesced read
  __syncthreads();
#pragma unroll
  for (int k = 0; k < 32; ++k) {
    int o = k * 256 + tid;                                 // block-linear out index
    int v = o >> 5, c = o & 31;
    out[(size_t)(v0 + v) * 32 + c] = t[v][c];              // coalesced write
  }
}

// bf16 [V][32] -> fp32 NCDHW [32][V]
__global__ void __launch_bounds__(256)
k_vc32_to_nchw(const bf16_t* __restrict__ in, float* __restrict__ out, int V) {
  __shared__ bf16_t t[256][33];
  int v0 = blockIdx.x * 256;
  int tid = threadIdx.x;
#pragma unroll
  for (int k = 0; k < 32; ++k) {
    int o = k * 256 + tid;
    int v = o >> 5, c = o & 31;
    t[v][c] = in[(size_t)(v0 + v) * 32 + c];               // coalesced read
  }
  __syncthreads();
#pragma unroll
  for (int c = 0; c < 32; ++c)
    out[(size_t)c * V + v0 + tid] = (float)t[tid][c];      // coalesced write
}

__global__ void k_zero_bf16(bf16_t* __restrict__ p, int n) {
  int i = blockIdx.x * blockDim.x + threadIdx.x;
  if (i < n) p[i] = (bf16_t)0.0f;
}

// ---------------- weight packing ----------------
// OIDHW fp32 -> bf16 B-fragment order:
// packed[(((t*chunks + c)*ntiles + nt)*32 + lane)*16 + e]
//   lane: n = lane&15, half = lane>>4 ; element e <-> K = (e&7) + 8*half + 16*(e>>3)
//   cin = c*32 + K ; cout = nt*16 + n ; zero-padded out of range
//   (the zero padding is what makes unpredicated A-tail loads safe).
__global__ void k_pack_w(const float* __restrict__ w, bf16_t* __restrict__ wp,
                         int T, int Cin, int Cout, int chunks, int ntiles, int total) {
  int idx = blockIdx.x * blockDim.x + threadIdx.x;
  if (idx >= total) return;
  int e    = idx & 15;
  int lane = (idx >> 4) & 31;
  int rest = idx >> 9;
  int nt = rest % ntiles; rest /= ntiles;
  int c  = rest % chunks;
  int t  = rest / chunks;
  int n = lane & 15, half = lane >> 4;
  int k = (e & 7) + (half << 3) + ((e >> 3) << 4);
  int cin  = (c << 5) + k;
  int cout = (nt << 4) + n;
  float v = 0.0f;
  if (cin < Cin && cout < Cout) v = w[((size_t)cout * Cin + cin) * T + t];
  wp[idx] = (bf16_t)v;
}

// ---------------- implicit-GEMM conv3d via WMMA ----------------
// MODE: 0 = stride-1 conv, 1 = stride-2 conv, 2 = transpose-conv (x2 upsample)
template<int MODE>
__device__ __forceinline__ bool mapc(int o, int d, int pad, int insz, int& i) {
  if (MODE == 2) {
    int num = o + d - 1;            // lhs_dilation=2, pad=(1,2): i=(o+t-1)/2 when even
    i = num >> 1;
    return ((num & 1) == 0) & ((unsigned)i < (unsigned)insz);
  } else if (MODE == 1) {
    i = o * 2 + d - pad;
  } else {
    i = o + d - pad;
  }
  return (unsigned)i < (unsigned)insz;
}

template<int KS, int MODE, int CHUNKS, int NT, int MT>
__global__ void __launch_bounds__(256)
k_conv(const bf16_t* __restrict__ in, const bf16_t* __restrict__ wpk,
       const float* __restrict__ bias, bf16_t* __restrict__ out,
       const bf16_t* __restrict__ resid, const bf16_t* __restrict__ zpg,
       int inD, int inH, int inW, int outH, int outW,
       int Cin, int Cout, int outCs, int outCo, int relu, int mtiles) {
  int wave = (blockIdx.x * 256 + (int)threadIdx.x) >> 5;
  if (wave >= mtiles) return;       // wave-uniform: EXEC stays all-1s for WMMA
  int lane = threadIdx.x & 31;
  int m = lane & 15, half = lane >> 4;
  const int pad = KS >> 1;
  int hw = outH * outW;

  int oz[MT], oy[MT], ox[MT];
#pragma unroll
  for (int mt = 0; mt < MT; ++mt) {
    int vid = wave * (16 * MT) + mt * 16 + m;  // A-row voxel for this lane
    int z = vid / hw;
    int rr = vid - z * hw;
    int y = rr / outW;
    oz[mt] = z; oy[mt] = y; ox[mt] = rr - y * outW;
  }

  v8f zero8 = {};
  v8f acc[MT * NT];
#pragma unroll
  for (int i = 0; i < MT * NT; ++i) acc[i] = zero8;

#pragma unroll 1
  for (int dz = 0; dz < KS; ++dz) {
#pragma unroll
    for (int dy = 0; dy < KS; ++dy) {
#pragma unroll
      for (int dx = 0; dx < KS; ++dx) {
        int t = (dz * KS + dy) * KS + dx;
        // Branchless per-lane source pointer: real voxel or zero page.
        const bf16_t* ip[MT];
#pragma unroll
        for (int mt = 0; mt < MT; ++mt) {
          int iz, iy, ix;
          bool ok = mapc<MODE>(oz[mt], dz, pad, inD, iz) &
                    mapc<MODE>(oy[mt], dy, pad, inH, iy) &
                    mapc<MODE>(ox[mt], dx, pad, inW, ix);
          long long ofs = (((long long)iz * inH + iy) * inW + ix) * (long long)Cin;
          ip[mt] = ok ? (in + ofs) : zpg;
        }
#pragma unroll
        for (int c = 0; c < CHUNKS; ++c) {
          // A fragment: e 0..7 <-> K = 8*half+0..7 ; e 8..15 <-> K = 16+8*half+0..7
          // Unconditional loads: channel tail garbage multiplies zero-padded B.
          v16bf A[MT];
#pragma unroll
          for (int mt = 0; mt < MT; ++mt) {
            v8bf a0 = *(const v8bf*)(ip[mt] + (c << 5) + (half << 3));
            v8bf a1 = *(const v8bf*)(ip[mt] + (c << 5) + (half << 3) + 16);
#pragma unroll
            for (int e = 0; e < 8; ++e) { A[mt][e] = a0[e]; A[mt][e + 8] = a1[e]; }
          }
          // B fragments: packed contiguous; shared across the MT row-tiles.
          const bf16_t* bp = wpk + (((size_t)(t * CHUNKS + c) * NT) * 32 + lane) * 16;
#pragma unroll
          for (int nt = 0; nt < NT; ++nt) {
            v8bf b0 = *(const v8bf*)(bp + nt * 512);
            v8bf b1 = *(const v8bf*)(bp + nt * 512 + 8);
            v16bf B;
#pragma unroll
            for (int e = 0; e < 8; ++e) { B[e] = b0[e]; B[e + 8] = b1[e]; }
#pragma unroll
            for (int mt = 0; mt < MT; ++mt)
              acc[mt * NT + nt] = __builtin_amdgcn_wmma_f32_16x16x32_bf16(
                  false, A[mt], false, B, (short)0, acc[mt * NT + nt], false, false);
          }
        }
      }
    }
  }

  // Epilogue: D layout row M = r + 8*half, col n = lane&15.
  // relu folded into a single v_max; resid test hoisted (wave-uniform).
  int n = lane & 15;
  const float lobnd = relu ? 0.0f : -3.0e38f;
  if (resid != nullptr) {
#pragma unroll
    for (int mt = 0; mt < MT; ++mt) {
#pragma unroll
      for (int nt = 0; nt < NT; ++nt) {
        int ng = (nt << 4) + n;
        if (ng < Cout) {
          float bv = bias[ng];
#pragma unroll
          for (int r = 0; r < 8; ++r) {
            int mv = wave * (16 * MT) + mt * 16 + r + (half << 3);
            size_t oidx = (size_t)mv * outCs + outCo + ng;
            float v = acc[mt * NT + nt][r] + bv;
            v = v > lobnd ? v : lobnd;
            v += (float)resid[oidx];
            out[oidx] = (bf16_t)v;
          }
        }
      }
    }
  } else {
#pragma unroll
    for (int mt = 0; mt < MT; ++mt) {
#pragma unroll
      for (int nt = 0; nt < NT; ++nt) {
        int ng = (nt << 4) + n;
        if (ng < Cout) {
          float bv = bias[ng];
#pragma unroll
          for (int r = 0; r < 8; ++r) {
            int mv = wave * (16 * MT) + mt * 16 + r + (half << 3);
            size_t oidx = (size_t)mv * outCs + outCo + ng;
            float v = acc[mt * NT + nt][r] + bv;
            v = v > lobnd ? v : lobnd;
            out[oidx] = (bf16_t)v;
          }
        }
      }
    }
  }
}

// ---------------- host orchestration ----------------
extern "C" void kernel_launch(void* const* d_in, const int* in_sizes, int n_in,
                              void* d_out, int out_size, void* d_ws, size_t ws_size,
                              hipStream_t stream) {
  (void)in_sizes; (void)n_in; (void)out_size; (void)ws_size;
  char* ws = (char*)d_ws;
  size_t off = 0;
  auto alloc = [&](size_t bytes) -> void* {
    void* p = ws + off;
    off = (off + bytes + 255) & ~(size_t)255;
    return p;
  };

  // zero page for OOB / parity-miss taps (zeroed every call below)
  bf16_t* zpg = (bf16_t*)alloc(4096);

  auto pack = [&](int widx, int Cin, int Cout, int KS) -> bf16_t* {
    int T = KS * KS * KS;
    int chunks = (Cin + 31) / 32, ntiles = (Cout + 15) / 16;
    int total = T * chunks * ntiles * 512;
    bf16_t* wp = (bf16_t*)alloc((size_t)total * sizeof(bf16_t));
    k_pack_w<<<(total + 255) / 256, 256, 0, stream>>>(
        (const float*)d_in[widx], wp, T, Cin, Cout, chunks, ntiles, total);
    return wp;
  };
  auto conv = [&](const bf16_t* in, bf16_t* wp, int bidx, bf16_t* out, const bf16_t* resid,
                  int inD, int outD, int Cin, int Cout, int outCs, int outCo,
                  int KS, int mode, int relu) {
    int vox = outD * outD * outD;                 // all grids cubic, divisible by 32
    int chunks = (Cin + 31) / 32, nt = (Cout + 15) / 16;
    const float* b = (const float*)d_in[bidx];
#define LNCH(ks, md, ch, ntt, mt)                                               \
    do {                                                                        \
      int mtiles = vox / (16 * (mt));                                           \
      int blocks = (mtiles + 7) / 8; /* 8 waves / 256-thread block */           \
      k_conv<ks, md, ch, ntt, mt><<<blocks, 256, 0, stream>>>(                  \
          in, wp, b, out, resid, zpg, inD, inD, inD, outD, outD,                \
          Cin, Cout, outCs, outCo, relu, mtiles);                               \
    } while (0)
    if (KS == 1) {
      if (chunks == 2)      LNCH(1, 0, 2, 1, 1);
      else if (nt == 2)     LNCH(1, 0, 1, 2, 1);
      else                  LNCH(1, 0, 1, 1, 1);
    } else if (mode == 0) {
      if (chunks == 2) {
        if (nt == 4)        LNCH(3, 0, 2, 4, 2);
        else                LNCH(3, 0, 2, 1, 1);
      } else {
        if (nt == 4)        LNCH(3, 0, 1, 4, 2);
        else if (nt == 2)   LNCH(3, 0, 1, 2, 1);
        else                LNCH(3, 0, 1, 1, 1);
      }
    } else if (mode == 1) {
      LNCH(3, 1, 2, 4, 2);
    } else {
      if (chunks == 2) {
        if (nt == 4)        LNCH(3, 2, 2, 4, 2);
        else                LNCH(3, 2, 2, 2, 1);
      } else {
        LNCH(3, 2, 1, 2, 1);
      }
    }
#undef LNCH
  };

  struct IncW { bf16_t* w[5]; int b[5]; };
  auto packInc = [&](int base, int c) -> IncW {
    IncW r; int q = c / 4, h = c / 2;
    r.w[0] = pack(base + 0, c, q, 1); r.b[0] = base + 1;
    r.w[1] = pack(base + 2, q, q, 3); r.b[1] = base + 3;
    r.w[2] = pack(base + 4, q, h, 1); r.b[2] = base + 5;
    r.w[3] = pack(base + 6, c, q, 3); r.b[3] = base + 7;
    r.w[4] = pack(base + 8, q, h, 3); r.b[4] = base + 9;
    return r;
  };
  auto runInc = [&](const bf16_t* x, int D, int c, const IncW& W,
                    bf16_t* ta, bf16_t* tb, bf16_t* outB) {
    int q = c / 4, h = c / 2;
    conv(x,  W.w[0], W.b[0], ta,   nullptr, D, D, c, q, q, 0, 1, 0, 1);
    conv(ta, W.w[1], W.b[1], tb,   nullptr, D, D, q, q, q, 0, 3, 0, 1);
    conv(tb, W.w[2], W.b[2], outB, x,       D, D, q, h, c, 0, 1, 0, 1); // concat lo + x
    conv(x,  W.w[3], W.b[3], ta,   nullptr, D, D, c, q, q, 0, 3, 0, 1);
    conv(ta, W.w[4], W.b[4], outB, x,       D, D, q, h, c, h, 3, 0, 1); // concat hi + x
  };

  // ---- pack all weights (d_in index map: x=0, then (w,b) pairs in dict order) ----
  bf16_t* Wc0 = pack(1, 32, 64, 3);
  bf16_t* Wd0 = pack(3, 64, 64, 3);
  IncW R0w  = packInc(5, 64);
  bf16_t* Wc1 = pack(15, 64, 64, 3);
  bf16_t* Wd1 = pack(17, 64, 64, 3);
  IncW R1w  = packInc(19, 64);
  bf16_t* Wc2 = pack(29, 64, 64, 3);
  bf16_t* Wd2 = pack(31, 64, 64, 3);
  IncW R2w  = packInc(33, 64);
  bf16_t* Wmid = pack(43, 64, 64, 3);
  bf16_t* Wup2 = pack(45, 64, 64, 3);
  bf16_t* Wuc2 = pack(47, 64, 64, 3);
  IncW UR2w = packInc(49, 64);
  bf16_t* Wup1 = pack(59, 64, 32, 3);   // (O=32, I=64)
  bf16_t* Wuc1 = pack(61, 32, 32, 3);
  IncW UR1w = packInc(63, 32);
  bf16_t* Wup0 = pack(73, 32, 32, 3);
  bf16_t* Wuc0 = pack(75, 32, 32, 3);

  // ---- activation buffers (bf16, channels-last, +pad for unpredicated tail loads) ----
  const int V64 = 64 * 64 * 64, V32 = 32 * 32 * 32, V16 = 16 * 16 * 16, V8 = 8 * 8 * 8;
  auto abuf = [&](size_t elems) { return (bf16_t*)alloc(elems * sizeof(bf16_t) + 256); };
  bf16_t* XIN = abuf((size_t)V64 * 32);
  bf16_t* A0  = abuf((size_t)V64 * 64);
  bf16_t* B0  = abuf((size_t)V32 * 64);
  bf16_t* TA  = abuf((size_t)V32 * 16);
  bf16_t* TB  = abuf((size_t)V32 * 16);
  bf16_t* R0  = abuf((size_t)V32 * 64);
  bf16_t* A1  = abuf((size_t)V32 * 64);
  bf16_t* B1  = abuf((size_t)V16 * 64);
  bf16_t* R1  = abuf((size_t)V16 * 64);
  bf16_t* A2  = abuf((size_t)V16 * 64);
  bf16_t* B2  = abuf((size_t)V8 * 64);
  bf16_t* R2  = abuf((size_t)V8 * 64);
  bf16_t* MID = abuf((size_t)V8 * 64);
  bf16_t* U2  = abuf((size_t)V16 * 64);
  bf16_t* V2b = abuf((size_t)V16 * 64);
  bf16_t* W2  = abuf((size_t)V16 * 64);
  bf16_t* U1  = abuf((size_t)V32 * 32);
  bf16_t* V1b = abuf((size_t)V32 * 32);
  bf16_t* W1  = abuf((size_t)V32 * 32);
  bf16_t* U0  = A0;    // A0 free after down0; 64^3 x 32 fits
  bf16_t* FIN = XIN;   // XIN free after conv0

  // ---- network ----
  k_zero_bf16<<<8, 256, 0, stream>>>(zpg, 2048);
  k_nchw_to_vc32<<<V64 / 256, 256, 0, stream>>>((const float*)d_in[0], XIN, V64);

  conv(XIN, Wc0, 2, A0, nullptr, 64, 64, 32, 64, 64, 0, 3, 0, 0);
  conv(A0,  Wd0, 4, B0, nullptr, 64, 32, 64, 64, 64, 0, 3, 1, 1);
  runInc(B0, 32, 64, R0w, TA, TB, R0);

  conv(R0, Wc1, 16, A1, nullptr, 32, 32, 64, 64, 64, 0, 3, 0, 0);
  conv(A1, Wd1, 18, B1, nullptr, 32, 16, 64, 64, 64, 0, 3, 1, 1);
  runInc(B1, 16, 64, R1w, TA, TB, R1);

  conv(R1, Wc2, 30, A2, nullptr, 16, 16, 64, 64, 64, 0, 3, 0, 0);
  conv(A2, Wd2, 32, B2, nullptr, 16, 8, 64, 64, 64, 0, 3, 1, 1);
  runInc(B2, 8, 64, R2w, TA, TB, R2);

  conv(R2,  Wmid, 44, MID, nullptr, 8, 8, 64, 64, 64, 0, 3, 0, 0);

  conv(MID, Wup2, 46, U2,  nullptr, 8, 16, 64, 64, 64, 0, 3, 2, 0);
  conv(U2,  Wuc2, 48, V2b, nullptr, 16, 16, 64, 64, 64, 0, 3, 0, 1);
  runInc(V2b, 16, 64, UR2w, TA, TB, W2);

  conv(W2, Wup1, 60, U1,  nullptr, 16, 32, 64, 32, 32, 0, 3, 2, 0);
  conv(U1, Wuc1, 62, V1b, nullptr, 32, 32, 32, 32, 32, 0, 3, 0, 1);
  runInc(V1b, 32, 32, UR1w, TA, TB, W1);

  conv(W1, Wup0, 74, U0,  nullptr, 32, 64, 32, 32, 32, 0, 3, 2, 0);
  conv(U0, Wuc0, 76, FIN, nullptr, 64, 64, 32, 32, 32, 0, 3, 0, 0);

  k_vc32_to_nchw<<<V64 / 256, 256, 0, stream>>>(FIN, (float*)d_out, 32 * 0 + V64);
}
